// SeqBoatLayer_1821066133757
// MI455X (gfx1250) — matrix operand order repaired and checked by reading, not compiled, not run on hardware
//
#include <hip/hip_runtime.h>

#define DEV __device__ __forceinline__

constexpr int L = 8192;
constexpr int B = 4;
constexpr int D = 512;
constexpr int NDIM = 16;
constexpr int Z = 128;
constexpr int W = 128;
constexpr int NW = L / W;           // 64 windows per batch
constexpr int MP = 256;
constexpr float NEGV = -50000.0f;
constexpr int CCH = 512;            // scan chunk length
constexpr int NCH = L / CCH;        // 16 chunks
constexpr int MROWS = B * L;        // 32768 compacted rows
constexpr int LP = L + W;           // padded length for v-transpose

typedef __attribute__((ext_vector_type(16))) __bf16 v16bf;
typedef __attribute__((ext_vector_type(8)))  __bf16 bf16x8;
typedef __attribute__((ext_vector_type(8)))  float  v8f;

DEV v8f wmma_bf16(v16bf a, v16bf b, v8f c){
#if defined(__HIP_DEVICE_COMPILE__)
  return __builtin_amdgcn_wmma_f32_16x16x32_bf16(false, a, false, b, (short)0, c, false, false);
#else
  return c;
#endif
}

DEV float frcp(float x){
#if defined(__HIP_DEVICE_COMPILE__)
  return __builtin_amdgcn_rcpf(x);    // single v_rcp_f32
#else
  return 1.f / x;
#endif
}

DEV float sigm(float x){ return frcp(1.f + __expf(-x)); }
DEV float silu(float x){ return x * frcp(1.f + __expf(-x)); }

// A fragment: row-major A[M,K], 16x32 tile. Lane l: m = l&15, half = l>>4.
// Lane holds K = half*16 + 0..15 (contiguous), matching the B fragment.
DEV v16bf load_afrag(const __bf16* A, int lda, int mbase, int kbase, int lane){
  int m = mbase + (lane & 15);
  int half = lane >> 4;
  const __bf16* p = A + (size_t)m * lda + kbase + half * 16;
  bf16x8 lo = *(const bf16x8*)p;
  bf16x8 hi = *(const bf16x8*)(p + 8);
  v16bf a;
#pragma unroll
  for (int j = 0; j < 8; ++j){ a[j] = lo[j]; a[8 + j] = hi[j]; }
  return a;
}

// B fragment from transposed storage Bt[N,K] row-major, 32x16 (KxN) tile.
// Lane l: n = l&15, holds K = half*16 + 0..15 (contiguous in Bt row).
DEV v16bf load_bfrag(const __bf16* Bt, int ldb, int nbase, int kbase, int lane){
  int n = nbase + (lane & 15);
  int half = lane >> 4;
  const __bf16* p = Bt + (size_t)n * ldb + kbase + half * 16;
  bf16x8 lo = *(const bf16x8*)p;
  bf16x8 hi = *(const bf16x8*)(p + 8);
  v16bf b;
#pragma unroll
  for (int j = 0; j < 8; ++j){ b[j] = lo[j]; b[8 + j] = hi[j]; }
  return b;
}

DEV v16bf load_bfrag_guard(const __bf16* Bt, int ldb, int nbase, int kbase, int lane){
  int n = nbase + (lane & 15);
  int half = lane >> 4;
  v16bf b;
  if (n >= 0){
    const __bf16* p = Bt + (size_t)n * ldb + kbase + half * 16;
    bf16x8 lo = *(const bf16x8*)p;
    bf16x8 hi = *(const bf16x8*)(p + 8);
#pragma unroll
    for (int j = 0; j < 8; ++j){ b[j] = lo[j]; b[8 + j] = hi[j]; }
  } else {
#pragma unroll
    for (int j = 0; j < 16; ++j) b[j] = (__bf16)0.f;
  }
  return b;
}

// ---------------- small prep kernels ----------------
__global__ void k_prep(const float* delta, const float* alpha, const float* beta,
                       const float* gamma, float* qdn, float* lqdn, float* cdn){
  int i = blockIdx.x * blockDim.x + threadIdx.x;
  if (i >= D * NDIM) return;
  float p = sigm(delta[i]);
  float a = sigm(alpha[i]);
  float pa = p * a;
  qdn[i]  = 1.f - pa;
  lqdn[i] = log1pf(-pa);
  cdn[i]  = p * beta[i] * gamma[i] * 0.25f;   // 1/sqrt(NDIM)
}

__global__ void k_wzt(const float* Wz, __bf16* wzt){
  int i = blockIdx.x * blockDim.x + threadIdx.x;   // D*Z
  if (i >= D * Z) return;
  int d = i / Z, z = i % Z;
  wzt[(size_t)z * D + d] = (__bf16)Wz[i];
}

__global__ void k_wvt(const float* Wv, __bf16* wvt){
  int i = blockIdx.x * blockDim.x + threadIdx.x;   // D*D
  if (i >= D * D) return;
  int r = i / D, c = i % D;
  wvt[(size_t)c * D + r] = (__bf16)Wv[i];
}

__global__ void k_vtzero(__bf16* vt){
  int i = blockIdx.x * blockDim.x + threadIdx.x;   // B*D*W pad entries
  if (i >= B * D * W) return;
  int w = i % W; int bd = i / W;
  vt[(size_t)bd * LP + w] = (__bf16)0.f;
}

// ---------------- chunked linear-recurrence scan ----------------
__global__ void k_scanA(const float* __restrict__ x, const float* __restrict__ qdn, float* st){
  int i = blockIdx.x * blockDim.x + threadIdx.x;   // NCH*B*D
  if (i >= NCH * B * D) return;
  int d = i % D; int b = (i / D) % B; int ch = i / (D * B);
  float q[NDIM], h[NDIM];
#pragma unroll
  for (int n = 0; n < NDIM; ++n){ q[n] = qdn[d * NDIM + n]; h[n] = 0.f; }
  const float* xp = x + ((size_t)(ch * CCH) * B + b) * D + d;
  for (int l = 0; l < CCH; ++l){
    float xv = xp[(size_t)l * B * D];
#pragma unroll
    for (int n = 0; n < NDIM; ++n) h[n] = fmaf(q[n], h[n], xv);
  }
  float* sp = st + ((size_t)(ch * B + b) * D + d) * NDIM;
#pragma unroll
  for (int n = 0; n < NDIM; ++n) sp[n] = h[n];
}

__global__ void k_comb(const float* __restrict__ lqdn, float* st){
  int i = blockIdx.x * blockDim.x + threadIdx.x;   // B*D
  if (i >= B * D) return;
  int d = i % D; int b = i / D;
  float qc[NDIM], run[NDIM];
#pragma unroll
  for (int n = 0; n < NDIM; ++n){ qc[n] = __expf(lqdn[d * NDIM + n] * (float)CCH); run[n] = 0.f; }
  for (int ch = 0; ch < NCH; ++ch){
    float* sp = st + ((size_t)(ch * B + b) * D + d) * NDIM;
#pragma unroll
    for (int n = 0; n < NDIM; ++n){
      float le = sp[n];
      sp[n] = run[n];                       // true initial state for this chunk
      run[n] = fmaf(qc[n], run[n], le);     // true end state
    }
  }
}

__global__ void k_scanB(const float* __restrict__ x, const float* __restrict__ omega,
                        const float* __restrict__ qdn, const float* __restrict__ cdn,
                        const float* __restrict__ st, float* __restrict__ mx){
  int i = blockIdx.x * blockDim.x + threadIdx.x;   // NCH*B*D
  if (i >= NCH * B * D) return;
  int d = i % D; int b = (i / D) % B; int ch = i / (D * B);
  float q[NDIM], c[NDIM], h[NDIM];
  const float* sp = st + ((size_t)(ch * B + b) * D + d) * NDIM;
#pragma unroll
  for (int n = 0; n < NDIM; ++n){ q[n] = qdn[d*NDIM+n]; c[n] = cdn[d*NDIM+n]; h[n] = sp[n]; }
  float om = omega[d];
  const float* xp = x + ((size_t)(ch * CCH) * B + b) * D + d;
  float* mp = mx + ((size_t)b * L + ch * CCH) * D + d;
  for (int l = 0; l < CCH; ++l){
    float xv = xp[(size_t)l * B * D];
    float conv = 0.f;
#pragma unroll
    for (int n = 0; n < NDIM; ++n){ h[n] = fmaf(q[n], h[n], xv); conv = fmaf(c[n], h[n], conv); }
    mp[(size_t)l * D] = silu(conv + xv * om);
  }
}

// ---------------- gate, cumsum compaction ----------------
__global__ void k_pi(const float* __restrict__ mx, const float* __restrict__ Wp,
                     const float* bp, const float* temp, float* pi){
  int gid = blockIdx.x * blockDim.x + threadIdx.x;
  int row = gid >> 5; int lane = gid & 31;
  if (row >= MROWS) return;
  const float* r = mx + (size_t)row * D;
  float s = 0.f;
  for (int j = lane; j < D; j += 32) s = fmaf(r[j], Wp[j], s);
#pragma unroll
  for (int m = 16; m >= 1; m >>= 1) s += __shfl_xor(s, m, 32);
  if (lane == 0){
    float logit = (s + bp[0]) * __expf(-temp[0]);
    pi[row] = sigm(logit);
  }
}

__global__ void k_sel(const float* __restrict__ pi, int* idx, int* src, int* cnt){
  int b = blockIdx.x; int tid = threadIdx.x;       // 1024 threads, 8 elems each
  __shared__ int sd[1024];
  int sel[8]; int lsum = 0;
  int l0 = tid * 8;
#pragma unroll
  for (int j = 0; j < 8; ++j){ sel[j] = pi[(size_t)b * L + l0 + j] > 0.5f ? 1 : 0; lsum += sel[j]; }
  sd[tid] = lsum; __syncthreads();
  for (int off = 1; off < 1024; off <<= 1){
    int v = (tid >= off) ? sd[tid - off] : 0;
    __syncthreads();
    sd[tid] += v;
    __syncthreads();
  }
  int run = sd[tid] - lsum;
#pragma unroll
  for (int j = 0; j < 8; ++j){
    int l = l0 + j;
    run += sel[j];
    idx[(size_t)b * L + l] = sel[j] ? run : 0;
    if (sel[j]) src[(size_t)b * L + run - 1] = l;
  }
  if (tid == 1023) cnt[b] = sd[1023];
}

__global__ void k_buildc(const float* __restrict__ mx, const int* __restrict__ src,
                         const int* __restrict__ cnt, __bf16* cb){
  int blk = blockIdx.x;                            // B*L compacted rows
  int b = blk / L; int i = blk % L;
  bool valid = i < cnt[b];
  int row = valid ? src[(size_t)b * L + i] : 0;
  const float* mr = mx + ((size_t)b * L + row) * D;
  __bf16* cr = cb + ((size_t)b * L + i) * D;
  for (int j = threadIdx.x; j < D; j += 128)
    cr[j] = valid ? (__bf16)mr[j] : (__bf16)0.f;
}

// ---------------- z GEMM (16x64/wave) + silu + rotary -> q,k (bf16) ----------------
__global__ void k_zqk(const __bf16* __restrict__ cb, const __bf16* __restrict__ wzt,
                      const float* qg, const float* qbv, const float* kg, const float* kbv,
                      __bf16* qo, __bf16* ko){
  int lane = threadIdx.x & 31; int wv = threadIdx.x >> 5;
  int Mt = blockIdx.x * 8 + wv;                    // MROWS/16 tiles
  int n0 = blockIdx.y * 64;                        // Z/64 blocks
  v8f acc[4] = {};
  for (int k = 0; k < D; k += 32){
    v16bf a = load_afrag(cb, D, Mt * 16, k, lane);
#pragma unroll
    for (int nt = 0; nt < 4; ++nt){
      v16bf bb = load_bfrag(wzt, D, n0 + nt * 16, k, lane);
      acc[nt] = wmma_bf16(a, bb, acc[nt]);
    }
  }
  int half = lane >> 4;
#pragma unroll
  for (int nt = 0; nt < 4; ++nt){
    int ncol = n0 + nt * 16 + (lane & 15);
    int j = ncol >> 1;
    float freq = __powf(10000.f, -(float)(2 * j) / (float)Z);
    float sgn = (ncol & 1) ? 1.f : -1.f;
    float gq = qg[ncol], oq = qbv[ncol], gk = kg[ncol], ok = kbv[ncol];
#pragma unroll
    for (int r = 0; r < 8; ++r){
      int mrow = Mt * 16 + r + 8 * half;           // global compacted row
      int pos = mrow & (L - 1);                    // position within batch
      float z = silu(acc[nt][r]);
      float qv = z * gq + oq;
      float kv = z * gk + ok;
      float qp = __shfl_xor(qv, 1, 32);
      float kp = __shfl_xor(kv, 1, 32);
      float sn, cs; __sincosf((float)pos * freq, &sn, &cs);
      qo[(size_t)mrow * Z + ncol] = (__bf16)(qv * cs + qp * sgn * sn);
      ko[(size_t)mrow * Z + ncol] = (__bf16)(kv * cs + kp * sgn * sn);
    }
  }
}

// ---------------- v GEMM (16x64/wave) + silu, transposed store into padded vT ----------------
__global__ void k_v(const __bf16* __restrict__ cb, const __bf16* __restrict__ wvt,
                    __bf16* vt){
  int lane = threadIdx.x & 31; int wv = threadIdx.x >> 5;
  int Mt = blockIdx.x * 8 + wv;
  int n0 = blockIdx.y * 64;                        // D/64 blocks
  v8f acc[4] = {};
  for (int k = 0; k < D; k += 32){
    v16bf a = load_afrag(cb, D, Mt * 16, k, lane);
#pragma unroll
    for (int nt = 0; nt < 4; ++nt){
      v16bf bb = load_bfrag(wvt, D, n0 + nt * 16, k, lane);
      acc[nt] = wmma_bf16(a, bb, acc[nt]);
    }
  }
  int half = lane >> 4;
  int mrow0 = Mt * 16 + 8 * half;                  // 8 consecutive rows
  int b = mrow0 / L; int i0 = mrow0 % L;
#pragma unroll
  for (int nt = 0; nt < 4; ++nt){
    int dcol = n0 + nt * 16 + (lane & 15);
    bf16x8 pack;
#pragma unroll
    for (int r = 0; r < 8; ++r) pack[r] = (__bf16)silu(acc[nt][r]);
    *(bf16x8*)(vt + ((size_t)b * D + dcol) * LP + W + i0) = pack;
  }
}

// ---------------- windowed attention: scores + softmax (2-sweep) ----------------
__global__ void k_scores(const __bf16* __restrict__ qv, const __bf16* __restrict__ kv,
                         const float* __restrict__ relb, const int* __restrict__ cnt,
                         __bf16* __restrict__ P, float* __restrict__ rsum){
  int win = blockIdx.x;                            // B*NW
  int b = win / NW; int w = win % NW; int base = w * W;
  int lane = threadIdx.x & 31; int wave = threadIdx.x >> 5;
  int m0 = wave * 16;
  int half = lane >> 4;
  int count = cnt[b];
  const float rscale = 0.08838834764831845f;       // 1/sqrt(Z)

  v16bf af[4];
  int rowbase = b * L + base + m0;
#pragma unroll
  for (int kk = 0; kk < 4; ++kk) af[kk] = load_afrag(qv, Z, rowbase, kk * 32, lane);

  const __bf16* kvb = kv + (size_t)b * L * Z;

  auto tile_scores = [&](int t, float s8[8]){
    v8f acc = {};
    int nbase = base - W + t * 16;
#pragma unroll
    for (int kk = 0; kk < 4; ++kk){
      v16bf bb = load_bfrag_guard(kvb, Z, nbase, kk * 32, lane);
      acc = wmma_bf16(af[kk], bb, acc);
    }
    int kcol = t * 16 + (lane & 15);
    int kabs = base + kcol - W;
#pragma unroll
    for (int r = 0; r < 8; ++r){
      int m = m0 + r + 8 * half;
      int rel = kcol - m + 127;                    // (kcol-W) - m + MP-1
      rel = rel < 0 ? 0 : (rel > 2 * MP - 2 ? 2 * MP - 2 : rel);
      float sc = acc[r] * rscale + relb[rel];
      bool ok = (kcol - W <= m) && (kabs >= 0) && (kabs < count);
      s8[r] = ok ? sc : NEGV;
    }
  };

  float rmax[8];
#pragma unroll
  for (int r = 0; r < 8; ++r) rmax[r] = -3.0e38f;
  for (int t = 0; t < 16; ++t){
    float s8[8];
    tile_scores(t, s8);
#pragma unroll
    for (int r = 0; r < 8; ++r){
      float v = s8[r];
      v = fmaxf(v, __shfl_xor(v, 1, 32));
      v = fmaxf(v, __shfl_xor(v, 2, 32));
      v = fmaxf(v, __shfl_xor(v, 4, 32));
      v = fmaxf(v, __shfl_xor(v, 8, 32));
      rmax[r] = fmaxf(rmax[r], v);
    }
  }

  float rs[8];
#pragma unroll
  for (int r = 0; r < 8; ++r) rs[r] = 0.f;
  __bf16* Pw = P + (size_t)win * W * (2 * W);
  for (int t = 0; t < 16; ++t){
    float s8[8];
    tile_scores(t, s8);
    int kcol = t * 16 + (lane & 15);
#pragma unroll
    for (int r = 0; r < 8; ++r){
      float e = __expf(s8[r] - rmax[r]);
      rs[r] += e;
      int m = m0 + r + 8 * half;
      Pw[(size_t)m * (2 * W) + kcol] = (__bf16)e;
    }
  }
#pragma unroll
  for (int r = 0; r < 8; ++r){
    float v = rs[r];
    v += __shfl_xor(v, 1, 32);
    v += __shfl_xor(v, 2, 32);
    v += __shfl_xor(v, 4, 32);
    v += __shfl_xor(v, 8, 32);
    if ((lane & 15) == 0){
      int m = m0 + r + 8 * half;
      rsum[(size_t)b * L + base + m] = v;
    }
  }
}

// ---------------- h = (P @ vw) / rowsum, 16x64 per wave ----------------
__global__ void k_h(const __bf16* __restrict__ P, const __bf16* __restrict__ vt,
                    const float* __restrict__ rsum, float* __restrict__ h){
  int win = blockIdx.x;                            // B*NW
  int b = win / NW; int w = win % NW; int base = w * W;
  int n0 = blockIdx.y * 64;                        // D/64 blocks
  int lane = threadIdx.x & 31; int wave = threadIdx.x >> 5;
  const __bf16* Pw = P + (size_t)win * W * (2 * W);
  const __bf16* Bt = vt + (size_t)b * D * LP + base;  // padded: vw[k] = Bt[d*LP + k]
  v8f acc[4] = {};
#pragma unroll
  for (int k = 0; k < 2 * W; k += 32){
    v16bf a = load_afrag(Pw, 2 * W, wave * 16, k, lane);
#pragma unroll
    for (int nt = 0; nt < 4; ++nt){
      v16bf bb = load_bfrag(Bt, LP, n0 + nt * 16, k, lane);
      acc[nt] = wmma_bf16(a, bb, acc[nt]);
    }
  }
  int half = lane >> 4;
  float inv[8];
#pragma unroll
  for (int r = 0; r < 8; ++r){
    int row = base + wave * 16 + r + 8 * half;
    inv[r] = frcp(rsum[(size_t)b * L + row]);
  }
#pragma unroll
  for (int nt = 0; nt < 4; ++nt){
    int d = n0 + nt * 16 + (lane & 15);
#pragma unroll
    for (int r = 0; r < 8; ++r){
      int row = base + wave * 16 + r + 8 * half;
      h[((size_t)b * L + row) * D + d] = acc[nt][r] * inv[r];
    }
  }
}

// ---------------- gather back + residual ----------------
__global__ void k_final(const float* __restrict__ x, const float* __restrict__ h,
                        const float* __restrict__ pi, const int* __restrict__ idx,
                        float* __restrict__ out){
  int gid = blockIdx.x * blockDim.x + threadIdx.x; // L*B*D
  if (gid >= L * B * D) return;
  int d = gid % D;
  int b = (gid / D) % B;
  int l = gid / (D * B);
  int id = idx[(size_t)b * L + l];
  float add = 0.f;
  if (id > 0) add = h[((size_t)b * L + (id - 1)) * D + d] * pi[(size_t)b * L + l];
  out[gid] = x[gid] + add;
}

extern "C" void kernel_launch(void* const* d_in, const int* in_sizes, int n_in,
                              void* d_out, int out_size, void* d_ws, size_t ws_size,
                              hipStream_t stream) {
  const float* x     = (const float*)d_in[0];
  const float* delta = (const float*)d_in[1];
  const float* alpha = (const float*)d_in[2];
  const float* beta  = (const float*)d_in[3];
  const float* gamma = (const float*)d_in[4];
  const float* omega = (const float*)d_in[5];
  const float* Wp    = (const float*)d_in[6];
  const float* bp    = (const float*)d_in[7];
  const float* temp  = (const float*)d_in[8];
  const float* Wz    = (const float*)d_in[9];
  const float* qg    = (const float*)d_in[10];
  const float* qbv   = (const float*)d_in[11];
  const float* kg    = (const float*)d_in[12];
  const float* kbv   = (const float*)d_in[13];
  const float* Wv    = (const float*)d_in[14];
  const float* relb  = (const float*)d_in[15];
  float* out = (float*)d_out;

  char* ws = (char*)d_ws;
  size_t off = 0;
  auto take = [&](size_t bytes) -> char* {
    char* p = ws + off;
    off += (bytes + 255) & ~(size_t)255;
    return p;
  };
  float*  qdn  = (float*)take((size_t)D * NDIM * 4);
  float*  lqdn = (float*)take((size_t)D * NDIM * 4);
  float*  cdn  = (float*)take((size_t)D * NDIM * 4);
  float*  st   = (float*)take((size_t)NCH * B * D * NDIM * 4);
  float*  mx   = (float*)take((size_t)B * L * D * 4);   // aliased as h later
  float*  pi   = (float*)take((size_t)B * L * 4);
  int*    idx  = (int*)  take((size_t)B * L * 4);
  int*    src  = (int*)  take((size_t)B * L * 4);
  int*    cnt  = (int*)  take((size_t)B * 4);
  __bf16* cb   = (__bf16*)take((size_t)B * L * D * 2);
  __bf16* wzt  = (__bf16*)take((size_t)Z * D * 2);
  __bf16* wvt  = (__bf16*)take((size_t)D * D * 2);
  __bf16* qo   = (__bf16*)take((size_t)B * L * Z * 2);
  __bf16* ko   = (__bf16*)take((size_t)B * L * Z * 2);
  __bf16* vt   = (__bf16*)take((size_t)B * D * LP * 2);
  __bf16* P    = (__bf16*)take((size_t)B * NW * W * (2 * W) * 2);
  float*  rsum = (float*)take((size_t)B * L * 4);
  float*  hbuf = mx;                                    // alias: mx dead after k_buildc

  k_prep  <<<(D * NDIM + 255) / 256, 256, 0, stream>>>(delta, alpha, beta, gamma, qdn, lqdn, cdn);
  k_wzt   <<<(D * Z) / 256, 256, 0, stream>>>(Wz, wzt);
  k_wvt   <<<(D * D) / 256, 256, 0, stream>>>(Wv, wvt);
  k_vtzero<<<(B * D * W) / 256, 256, 0, stream>>>(vt);

  k_scanA <<<(NCH * B * D) / 256, 256, 0, stream>>>(x, qdn, st);
  k_comb  <<<(B * D) / 256, 256, 0, stream>>>(lqdn, st);
  k_scanB <<<(NCH * B * D) / 256, 256, 0, stream>>>(x, omega, qdn, cdn, st, mx);

  k_pi    <<<MROWS / 8, 256, 0, stream>>>(mx, Wp, bp, temp, pi);
  k_sel   <<<B, 1024, 0, stream>>>(pi, idx, src, cnt);
  k_buildc<<<B * L, 128, 0, stream>>>(mx, src, cnt, cb);

  dim3 gz(MROWS / 128, Z / 64);
  k_zqk   <<<gz, 256, 0, stream>>>(cb, wzt, qg, qbv, kg, kbv, qo, ko);
  dim3 gv(MROWS / 128, D / 64);
  k_v     <<<gv, 256, 0, stream>>>(cb, wvt, vt);

  k_scores<<<B * NW, 256, 0, stream>>>(qo, ko, relb, cnt, P, rsum);
  dim3 gh(B * NW, D / 64);
  k_h     <<<gh, 256, 0, stream>>>(P, vt, rsum, hbuf);

  k_final <<<(L * B * D) / 256, 256, 0, stream>>>(x, hbuf, pi, idx, out);
  (void)in_sizes; (void)n_in; (void)out_size; (void)ws_size;
}
